// STCOGAT_962072674788
// MI455X (gfx1250) — matrix operand order repaired and checked by reading, not compiled, not run on hardware
//
#include <hip/hip_runtime.h>
#include <math.h>
#include <stdint.h>

// Problem constants (match reference file)
#define NN   100000
#define EE   1600000
#define DIN  128
#define DOUT 64
// derived, all exact:
#define GEMM_BLOCKS   (NN / 16)     // 6250
#define LOGIT_BLOCKS  (EE / 256)    // 6250
#define SCAT_BLOCKS   (EE / 16)     // 100000

typedef __attribute__((ext_vector_type(16))) _Float16 v16h;
typedef __attribute__((ext_vector_type(8)))  float    v8f;

// ---------------------------------------------------------------------------
// Kernel 0: transpose + f16-convert the four weight matrices once:
// Wt[w][n][k] = (f16) W_w[k][n].   4 * 64 * 128 f16 = 64 KB, L2-resident.
// This makes every WMMA B-fragment a contiguous 32-byte run (2x b128 loads).
// ---------------------------------------------------------------------------
__global__ __launch_bounds__(256) void prep_weights(
    const float* __restrict__ Wq, const float* __restrict__ Wk,
    const float* __restrict__ Wv, const float* __restrict__ Ws,
    _Float16* __restrict__ Wt)
{
    const int i = blockIdx.x * 256 + threadIdx.x;  // 0 .. 4*DOUT*DIN-1
    const int w = i >> 13;                         // 8192 elems per matrix
    const int n = (i >> 7) & 63;                   // output column
    const int c = i & 127;                         // K index
    const float* W = (w == 0) ? Wq : (w == 1) ? Wk : (w == 2) ? Wv : Ws;
    Wt[i] = (_Float16)W[(size_t)c * DOUT + n];
}

// ---------------------------------------------------------------------------
// Kernel 1: fused q/k/v/skip GEMM.  One 16-row tile of x per block, staged to
// LDS as f16.  4 waves per block; wave w computes x_tile @ W_w + b_w with
// v_wmma_f32_16x16x32_f16 (4 K-steps x 4 N-tiles = 16 WMMA / wave, fully
// unrolled).  Skip result goes straight into d_out (scatter adds on top).
// ---------------------------------------------------------------------------
__global__ __launch_bounds__(128) void qkvs_gemm(
    const float* __restrict__ x,
    const _Float16* __restrict__ Wt,       // [4][DOUT][DIN] f16 (transposed)
    const float* __restrict__ bq, const float* __restrict__ bk,
    const float* __restrict__ bv, const float* __restrict__ bs,
    float* __restrict__ q, float* __restrict__ k, float* __restrict__ v,
    float* __restrict__ o)
{
    __shared__ _Float16 xs[16][DIN + 8];   // +8 f16 pad, keeps 16B alignment

    const int tid  = threadIdx.x;
    const int wave = tid >> 5;
    const int lane = tid & 31;
    const int row0 = blockIdx.x * 16;

    // Stage 16x128 fp32 tile -> f16 LDS via float4 loads (4 per thread);
    // compiler packs v_cvt_pk_f16_f32 + ds_store_b64.
    for (int i = tid; i < 16 * DIN / 4; i += 128) {
        const int r = i >> 5;              // 32 float4 chunks per row
        const int c = (i & 31) * 4;
        const float4 f = ((const float4*)(x + (size_t)(row0 + r) * DIN))[i & 31];
        xs[r][c + 0] = (_Float16)f.x;
        xs[r][c + 1] = (_Float16)f.y;
        xs[r][c + 2] = (_Float16)f.z;
        xs[r][c + 3] = (_Float16)f.w;
    }
    __syncthreads();

    const float* bias; float* out;
    switch (wave) {
        case 0:  bias = bq; out = q; break;
        case 1:  bias = bk; out = k; break;
        case 2:  bias = bv; out = v; break;
        default: bias = bs; out = o; break;
    }
    const _Float16* Wm = Wt + (size_t)wave * DOUT * DIN;

    const int m = lane & 15;   // M row (A) / N col (B,C,D)
    const int h = lane >> 4;   // lane half

    v8f acc[4] = {};           // 4 N-tiles of 16x16 f32

    #pragma unroll
    for (int kc = 0; kc < 4; ++kc) {
        const int kb = kc * 32;
        // A fragment, contiguous K order (matches B):
        // elem i -> K = kb + 16h + i
        v16h a;
        #pragma unroll
        for (int i = 0; i < 8; ++i) a[i]     = xs[m][kb + 16 * h + i];
        #pragma unroll
        for (int i = 0; i < 8; ++i) a[8 + i] = xs[m][kb + 16 * h + 8 + i];

        #pragma unroll
        for (int nt = 0; nt < 4; ++nt) {
            const int n0 = nt * 16;
            // B fragment (32x16): N = n0+m, elem i -> K = kb + 16h + i.
            // With transposed f16 weights this is one contiguous 32B run.
            const v16h b =
                *(const v16h*)(Wm + (size_t)(n0 + m) * DIN + kb + 16 * h);
            acc[nt] = __builtin_amdgcn_wmma_f32_16x16x32_f16(
                false, a, false, b, (short)0, acc[nt], false, false);
        }
    }

    // Epilogue: C/D layout is VGPR r -> M = r + 8h, N = m.  Add bias, store.
    #pragma unroll
    for (int nt = 0; nt < 4; ++nt) {
        const int n0 = nt * 16;
        const float bb = bias[n0 + m];
        #pragma unroll
        for (int r = 0; r < 8; ++r) {
            const int row = row0 + 8 * h + r;
            out[(size_t)row * DOUT + n0 + m] = acc[nt][r] + bb;
        }
    }
}

// ---------------------------------------------------------------------------
// Kernel 2: per-edge logits alpha[e] = dot(q[dst], k[src]) / sqrt(64),
// plus deterministic per-block partial sums of alpha and alpha^2.
// One edge per thread; q/k rows (256B) read as float4 (L2-resident).
// ---------------------------------------------------------------------------
__global__ __launch_bounds__(256) void edge_logits(
    const float* __restrict__ q, const float* __restrict__ k,
    const int* __restrict__ src, const int* __restrict__ dst,
    float* __restrict__ alpha,
    float* __restrict__ p1, float* __restrict__ p2)
{
    __shared__ float r1[256];
    __shared__ float r2[256];

    const int tid = threadIdx.x;
    const long long e = (long long)blockIdx.x * 256 + tid;   // E exact multiple

    const int s = src[e], d = dst[e];
    const float4* qr = (const float4*)(q + (size_t)d * DOUT);
    const float4* kr = (const float4*)(k + (size_t)s * DOUT);

    float acc = 0.0f;
    #pragma unroll
    for (int i = 0; i < DOUT / 4; ++i) {
        const float4 a4 = qr[i], b4 = kr[i];
        acc += a4.x * b4.x + a4.y * b4.y + a4.z * b4.z + a4.w * b4.w;
    }
    const float a = acc * 0.125f;   // 1/sqrt(64)
    alpha[e] = a;

    r1[tid] = a;
    r2[tid] = a * a;
    __syncthreads();
    for (int sft = 128; sft > 0; sft >>= 1) {
        if (tid < sft) { r1[tid] += r1[tid + sft]; r2[tid] += r2[tid + sft]; }
        __syncthreads();
    }
    if (tid == 0) { p1[blockIdx.x] = r1[0]; p2[blockIdx.x] = r2[0]; }
}

// ---------------------------------------------------------------------------
// Kernel 2b: single-block double-precision reduction of the partials ->
// stats[0] = mean, stats[1] = SCALE_PARAM / std_unbiased (ddof=1).
// ---------------------------------------------------------------------------
__global__ __launch_bounds__(256) void finalize_stats(
    const float* __restrict__ p1, const float* __restrict__ p2,
    int nparts, float* __restrict__ stats)
{
    __shared__ double s1s[256];
    __shared__ double s2s[256];

    const int tid = threadIdx.x;
    double s1 = 0.0, s2 = 0.0;
    for (int i = tid; i < nparts; i += 256) { s1 += (double)p1[i]; s2 += (double)p2[i]; }
    s1s[tid] = s1; s2s[tid] = s2;
    __syncthreads();
    for (int sft = 128; sft > 0; sft >>= 1) {
        if (tid < sft) { s1s[tid] += s1s[tid + sft]; s2s[tid] += s2s[tid + sft]; }
        __syncthreads();
    }
    if (tid == 0) {
        const double n    = (double)EE;
        const double mean = s1s[0] / n;
        double var = (s2s[0] - n * mean * mean) / (n - 1.0);   // unbiased
        if (var < 1e-30) var = 1e-30;
        stats[0] = (float)mean;
        stats[1] = (float)(2.0 / sqrt(var));   // SCALE_PARAM / std
    }
}

// ---------------------------------------------------------------------------
// Kernel 3: w = sigmoid((alpha - mean) * coef); scatter-add v[src]*w into
// out[dst].  16 lanes per edge, float4 per lane (contiguous 256B row),
// 4x global_atomic_add_f32 per lane.
// ---------------------------------------------------------------------------
__global__ __launch_bounds__(256) void scatter_msg(
    const float* __restrict__ v,
    const int* __restrict__ src, const int* __restrict__ dst,
    const float* __restrict__ alpha, const float* __restrict__ stats,
    float* __restrict__ out)
{
    const int tid = threadIdx.x;
    const int eg  = tid >> 4;    // edge within block (0..15)
    const int c4  = tid & 15;    // float4 column chunk
    const long long e = (long long)blockIdx.x * 16 + eg;   // E exact multiple

    const float mean = stats[0];
    const float coef = stats[1];
    const float a = (alpha[e] - mean) * coef;
    const float w = 1.0f / (1.0f + __expf(-a));

    const int s = src[e], d = dst[e];
    const float4 vv = ((const float4*)(v + (size_t)s * DOUT))[c4];
    float* op = out + (size_t)d * DOUT + (size_t)c4 * 4;
    atomicAdd(op + 0, vv.x * w);
    atomicAdd(op + 1, vv.y * w);
    atomicAdd(op + 2, vv.z * w);
    atomicAdd(op + 3, vv.w * w);
}

// ---------------------------------------------------------------------------
extern "C" void kernel_launch(void* const* d_in, const int* in_sizes, int n_in,
                              void* d_out, int out_size, void* d_ws, size_t ws_size,
                              hipStream_t stream)
{
    (void)in_sizes; (void)n_in; (void)out_size; (void)ws_size;

    const float* x  = (const float*)d_in[0];
    const int*   ei = (const int*)d_in[1];     // edge_index [2, E]
    const float* Wq = (const float*)d_in[2];
    const float* bq = (const float*)d_in[3];
    const float* Wk = (const float*)d_in[4];
    const float* bk = (const float*)d_in[5];
    const float* Wv = (const float*)d_in[6];
    const float* bv = (const float*)d_in[7];
    const float* Ws = (const float*)d_in[8];
    const float* bs = (const float*)d_in[9];
    const int* src = ei;        // edge_index[0]
    const int* dst = ei + EE;   // edge_index[1]
    float* out = (float*)d_out;

    // Workspace carve-up (q/k/v/alpha chunk sizes are multiples of 16B, Wt is rounded up to 256B; base is
    // hipMalloc-aligned, so float4/v16h accesses stay aligned).
    float* q     = (float*)d_ws;                 // NN*DOUT
    float* k     = q + (size_t)NN * DOUT;        // NN*DOUT
    float* v     = k + (size_t)NN * DOUT;        // NN*DOUT
    float* alpha = v + (size_t)NN * DOUT;        // EE
    float* p1    = alpha + EE;                   // LOGIT_BLOCKS
    float* p2    = p1 + LOGIT_BLOCKS;            // LOGIT_BLOCKS
    float* stats = p2 + LOGIT_BLOCKS;            // 2
    _Float16* Wt = (_Float16*)(((uintptr_t)(stats + 2) + 255) &
                               ~(uintptr_t)255); // [4][DOUT][DIN] f16

    prep_weights<<<(4 * DOUT * DIN) / 256, 256, 0, stream>>>(Wq, Wk, Wv, Ws, Wt);
    qkvs_gemm<<<GEMM_BLOCKS, 128, 0, stream>>>(x, Wt, bq, bk, bv, bs,
                                               q, k, v, out);
    edge_logits<<<LOGIT_BLOCKS, 256, 0, stream>>>(q, k, src, dst, alpha, p1, p2);
    finalize_stats<<<1, 256, 0, stream>>>(p1, p2, LOGIT_BLOCKS, stats);
    scatter_msg<<<SCAT_BLOCKS, 256, 0, stream>>>(v, src, dst, alpha, stats, out);
}
